// VectorQuantizer_70403103916639
// MI455X (gfx1250) — matrix operand order repaired and checked by reading, not compiled, not run on hardware
//
#include <hip/hip_runtime.h>
#include <math.h>

typedef __attribute__((ext_vector_type(16))) _Float16 v16h;
typedef __attribute__((ext_vector_type(8)))  float    v8f;

#define N_ROWS  65536
#define D_DIM   64
#define K_CODES 1024
#define QSIZE   (N_ROWS * D_DIM)      // 4194304
#define LOSS_POS QSIZE
#define PERP_POS (QSIZE + 1)
#define IDX_POS  (QSIZE + 2)

// Workspace layout (byte offsets):
//  [0, 131072)        packed f16 B fragments: 128 frags * 32 lanes * 16 halves
//  [131072, 135168)   e_norm: 1024 f32
//  [135168, 139264)   histogram: 1024 int
//  [139264, 139268)   loss accumulator: 1 f32
#define WS_EOFF  0
#define WS_ENORM 131072
#define WS_HIST  135168
#define WS_LOSS  139264

// Dynamic LDS: packed fragments (131072 B) + e_norm (4096 B)
#define SMEM_BYTES 135168

// ---------------------------------------------------------------------------
// Zero the workspace accumulators (ws is poisoned to 0xAA by the harness).
__global__ void vq_zero(float* lossAcc, int* hist) {
    int t = blockIdx.x * blockDim.x + threadIdx.x;
    if (t < K_CODES) hist[t] = 0;
    if (t == 0) lossAcc[0] = 0.0f;
}

// ---------------------------------------------------------------------------
// Pack E ([D=64, K=1024] row-major f32) into f16 WMMA B-fragments.
// Fragment f = ct*2 + kb covers B rows (contraction d) 32*kb..32*kb+31 and
// columns (codes) ct*16..ct*16+15.  B layout (16-bit, 32x16): lane L holds
// column N = L%16, halves h=0..7 hold K = 8*(L/16) + h, h=8..15 hold K = 16 + 8*(L/16) + (h-8).
// Stored lane-major: packed[(f*32 + lane)*16 + h].
__global__ void vq_pack(const float* __restrict__ E, _Float16* __restrict__ packed) {
    int tid  = blockIdx.x * blockDim.x + threadIdx.x;   // 65536 threads
    int h    = tid & 15;
    int lane = (tid >> 4) & 31;
    int f    = tid >> 9;
    int kb   = f & 1;
    int ct   = f >> 1;
    int g    = lane >> 4;
    int n    = ct * 16 + (lane & 15);
    int d    = 32 * kb + 8 * g + h + (h >= 8 ? 8 : 0);
    packed[tid] = (_Float16)E[d * K_CODES + n];
}

// ---------------------------------------------------------------------------
// Per-code squared norms ||e_k||^2.
__global__ void vq_enorm(const float* __restrict__ E, float* __restrict__ enorm) {
    int k = blockIdx.x * blockDim.x + threadIdx.x;      // 1024 threads
    float s = 0.0f;
    #pragma unroll 8
    for (int d = 0; d < D_DIM; ++d) {
        float v = E[d * K_CODES + k];
        s += v * v;
    }
    enorm[k] = s;
}

// ---------------------------------------------------------------------------
// Main kernel: one wave32 per 16-row tile, 8 waves per block.  The block
// first stages the packed f16 B fragments (128 KB) and e_norm (4 KB) into
// LDS (shared by all 8 waves -> 8x less L2 traffic, ds-latency hot loop).
// Then: WMMA f16 scores, per-row argmax of 2*S - ||e||^2 (x2-unrolled with
// independent accumulator chains), gather winning codes, write
// quantized/indices, accumulate loss + histogram.
__launch_bounds__(256)
__global__ void vq_main(const float*   __restrict__ x,
                        const float*   __restrict__ E,
                        const _Float16* __restrict__ packed,
                        const float*   __restrict__ enorm,
                        float*         __restrict__ out,
                        int*           __restrict__ hist,
                        float*         __restrict__ lossAcc) {
    extern __shared__ char smem[];
    _Float16* sE    = (_Float16*)smem;             // 131072 B
    float*    sNorm = (float*)(smem + 131072);     // 4096 B

    // ---- Cooperative staging: 8192 x 16B chunks of packed E + 1024 norms.
    {
        const float4* src = (const float4*)packed;
        float4*       dst = (float4*)sE;
        for (int i = threadIdx.x; i < 8192; i += 256) dst[i] = src[i];
        for (int i = threadIdx.x; i < K_CODES; i += 256) sNorm[i] = enorm[i];
    }
    __syncthreads();

    const int lane     = threadIdx.x & 31;
    const int wave     = (blockIdx.x * blockDim.x + threadIdx.x) >> 5;   // 0..4095
    const int row_base = wave << 4;
    const int g        = lane >> 4;       // half-wave group
    const int nl       = lane & 15;

    // ---- Build A fragments (16x32 f16 each) for K(d)=0..31 and 32..63.
    // A layout: lane L holds row M=L%16; halves 0..7 -> K=8g..8g+7,
    // halves 8..15 -> K=16+8g..23+8g (fragment-local).
    const float4* xr = (const float4*)(x + (size_t)(row_base + nl) * D_DIM);
    float4 p0 = xr[2*g],      p1 = xr[2*g + 1];     // d =  8g ..  8g+7
    float4 p2 = xr[4 + 2*g],  p3 = xr[5 + 2*g];     // d = 16+8g .. 23+8g
    float4 p4 = xr[8 + 2*g],  p5 = xr[9 + 2*g];     // d = 32+8g ..
    float4 p6 = xr[12 + 2*g], p7 = xr[13 + 2*g];    // d = 48+8g ..
    v16h a0, a1;
    a0[0]=(_Float16)p0.x; a0[1]=(_Float16)p0.y; a0[2]=(_Float16)p0.z; a0[3]=(_Float16)p0.w;
    a0[4]=(_Float16)p1.x; a0[5]=(_Float16)p1.y; a0[6]=(_Float16)p1.z; a0[7]=(_Float16)p1.w;
    a0[8]=(_Float16)p2.x; a0[9]=(_Float16)p2.y; a0[10]=(_Float16)p2.z; a0[11]=(_Float16)p2.w;
    a0[12]=(_Float16)p3.x; a0[13]=(_Float16)p3.y; a0[14]=(_Float16)p3.z; a0[15]=(_Float16)p3.w;
    a1[0]=(_Float16)p4.x; a1[1]=(_Float16)p4.y; a1[2]=(_Float16)p4.z; a1[3]=(_Float16)p4.w;
    a1[4]=(_Float16)p5.x; a1[5]=(_Float16)p5.y; a1[6]=(_Float16)p5.z; a1[7]=(_Float16)p5.w;
    a1[8]=(_Float16)p6.x; a1[9]=(_Float16)p6.y; a1[10]=(_Float16)p6.z; a1[11]=(_Float16)p6.w;
    a1[12]=(_Float16)p7.x; a1[13]=(_Float16)p7.y; a1[14]=(_Float16)p7.z; a1[15]=(_Float16)p7.w;

    float best[8];
    int   bidx[8];
    #pragma unroll
    for (int v = 0; v < 8; ++v) { best[v] = -3.4e38f; bidx[v] = 0; }

    const v16h*  bfrag = (const v16h*)sE;
    const float* enp   = sNorm + nl;

    for (int ct = 0; ct < 64; ct += 2) {
        // --- tile ct (chain X) and tile ct+1 (chain Y): fully independent.
        v16h bx0 = bfrag[(ct * 2 + 0) * 32 + lane];
        v16h bx1 = bfrag[(ct * 2 + 1) * 32 + lane];
        v16h by0 = bfrag[(ct * 2 + 2) * 32 + lane];
        v16h by1 = bfrag[(ct * 2 + 3) * 32 + lane];
        v8f cx = {}, cy = {};
        cx = __builtin_amdgcn_wmma_f32_16x16x32_f16(false, a0, false, bx0,
                                                    (short)0, cx, false, false);
        cy = __builtin_amdgcn_wmma_f32_16x16x32_f16(false, a0, false, by0,
                                                    (short)0, cy, false, false);
        cx = __builtin_amdgcn_wmma_f32_16x16x32_f16(false, a1, false, bx1,
                                                    (short)0, cx, false, false);
        cy = __builtin_amdgcn_wmma_f32_16x16x32_f16(false, a1, false, by1,
                                                    (short)0, cy, false, false);

        int   nx  = ct * 16 + nl;
        int   ny  = nx + 16;
        float enx = enp[ct * 16];
        float eny = enp[ct * 16 + 16];
        #pragma unroll
        for (int v = 0; v < 8; ++v) {
            float sx = 2.0f * cx[v] - enx;      // maximize <=> min distance
            if (sx > best[v]) { best[v] = sx; bidx[v] = nx; }
        }
        #pragma unroll
        for (int v = 0; v < 8; ++v) {
            float sy = 2.0f * cy[v] - eny;
            if (sy > best[v]) { best[v] = sy; bidx[v] = ny; }
        }
    }

    // ---- Argmax reduction across the 16 lanes holding each row
    // (xor masks < 16 stay within each half-wave group). Ties -> lower index.
    #pragma unroll
    for (int off = 8; off >= 1; off >>= 1) {
        #pragma unroll
        for (int v = 0; v < 8; ++v) {
            float ov = __shfl_xor(best[v], off, 32);
            int   oi = __shfl_xor(bidx[v], off, 32);
            if (ov > best[v] || (ov == best[v] && oi < bidx[v])) {
                best[v] = ov; bidx[v] = oi;
            }
        }
    }
    // Now every lane in group g holds indices for rows 8g..8g+7 (v=0..7).

    // ---- Indices + histogram (one writer per 16-lane group).
    if (nl == 0) {
        #pragma unroll
        for (int v = 0; v < 8; ++v) {
            int row = row_base + 8 * g + v;
            out[IDX_POS + row] = (float)bidx[v];
            atomicAdd(&hist[bidx[v]], 1);
        }
    }

    // ---- Gather winning codes, write quantized (== straight-through value),
    // accumulate squared error.  Lane handles d = 4*nl .. 4*nl+3 per row.
    float lsum = 0.0f;
    const int d0 = nl * 4;
    #pragma unroll
    for (int v = 0; v < 8; ++v) {
        int row = row_base + 8 * g + v;
        int idx = bidx[v];
        float4 xv = *(const float4*)(x + (size_t)row * D_DIM + d0);
        float4 q;
        q.x = E[(d0 + 0) * K_CODES + idx];
        q.y = E[(d0 + 1) * K_CODES + idx];
        q.z = E[(d0 + 2) * K_CODES + idx];
        q.w = E[(d0 + 3) * K_CODES + idx];
        float dx = q.x - xv.x, dy = q.y - xv.y, dz = q.z - xv.z, dw = q.w - xv.w;
        lsum += dx * dx + dy * dy + dz * dz + dw * dw;
        *(float4*)(out + (size_t)row * D_DIM + d0) = q;
    }
    #pragma unroll
    for (int off = 16; off >= 1; off >>= 1) lsum += __shfl_xor(lsum, off, 32);
    if (lane == 0) atomicAdd(lossAcc, lsum);
}

// ---------------------------------------------------------------------------
// Perplexity from histogram + final loss scaling.
__global__ void vq_finalize(const int* __restrict__ hist,
                            const float* __restrict__ lossAcc,
                            float* __restrict__ out) {
    __shared__ float red[256];
    int t = threadIdx.x;
    float s = 0.0f;
    for (int k = t; k < K_CODES; k += 256) {
        float p = (float)hist[k] * (1.0f / (float)N_ROWS);
        s += p * logf(p + 1e-10f);
    }
    red[t] = s;
    __syncthreads();
    for (int w = 128; w >= 1; w >>= 1) {
        if (t < w) red[t] += red[t + w];
        __syncthreads();
    }
    if (t == 0) {
        // loss = q_latent + 0.25*e_latent = 1.25 * mean((q-x)^2)
        out[LOSS_POS] = lossAcc[0] * (1.25f / (float)QSIZE);
        out[PERP_POS] = expf(-red[0]);
    }
}

// ---------------------------------------------------------------------------
extern "C" void kernel_launch(void* const* d_in, const int* in_sizes, int n_in,
                              void* d_out, int out_size, void* d_ws, size_t ws_size,
                              hipStream_t stream) {
    (void)in_sizes; (void)n_in; (void)out_size; (void)ws_size;
    const float* x = (const float*)d_in[0];     // [64,32,32,64] f32
    const float* E = (const float*)d_in[1];     // [64,1024] f32
    float* out = (float*)d_out;

    char* ws = (char*)d_ws;
    _Float16* packed  = (_Float16*)(ws + WS_EOFF);
    float*    enorm   = (float*)(ws + WS_ENORM);
    int*      hist    = (int*)(ws + WS_HIST);
    float*    lossAcc = (float*)(ws + WS_LOSS);

    vq_zero    <<<4,   256, 0,          stream>>>(lossAcc, hist);
    vq_pack    <<<256, 256, 0,          stream>>>(E, packed);
    vq_enorm   <<<4,   256, 0,          stream>>>(E, enorm);
    vq_main    <<<512, 256, SMEM_BYTES, stream>>>(x, E, packed, enorm, out, hist, lossAcc);
    vq_finalize<<<1,   256, 0,          stream>>>(hist, lossAcc, out);
}
